// SlotGenerator_30511447671270
// MI455X (gfx1250) — matrix operand order repaired and checked by reading, not compiled, not run on hardware
//
#include <hip/hip_runtime.h>

// ---------------- problem constants (from reference) ----------------
#define B_     4
#define S_     512
#define H_     768
#define V_     35000
#define J_     45
#define LSLOT_ 4
#define NG_    5
#define T_     5
#define N_     (B_ * J_)   // 180 rows
#define MP_    192          // N_ padded to multiple of 16 for WMMA
#define VP_    35008        // V_ padded to multiple of 16 for WMMA

typedef __attribute__((ext_vector_type(16))) __bf16 v16bf;
typedef __attribute__((ext_vector_type(8)))  __bf16 v8bf;
typedef __attribute__((ext_vector_type(8)))  float  v8f;

__device__ inline __bf16 f2bf(float f) {
  union { float f; unsigned u; } in; in.f = f;
  unsigned r = in.u + 0x7FFFu + ((in.u >> 16) & 1u);   // round-to-nearest-even
  unsigned short h = (unsigned short)(r >> 16);
  __bf16 out;
  __builtin_memcpy(&out, &h, 2);
  return out;
}

__device__ inline float sigmoidf_(float x) { return 1.0f / (1.0f + __expf(-x)); }

// ---------------- one-time: embed_W (f32, VxH) -> bf16 padded (VPxH) ----------------
__global__ void k_embed_bf16(const float* __restrict__ embed, __bf16* __restrict__ embB) {
  long long idx = (long long)blockIdx.x * blockDim.x + threadIdx.x;
  const long long total = (long long)VP_ * H_;
  if (idx >= total) return;
  int row = (int)(idx / H_);
  float v = (row < V_) ? embed[idx] : 0.0f;
  embB[idx] = f2bf(v);
}

// ---------------- one-time: build w0 = tile(slot embedding sums), h0 = repeat(hidden) ----------------
__global__ void k_init_state(const float* __restrict__ embed, const float* __restrict__ hidden,
                             const int* __restrict__ slot,
                             float* __restrict__ w, float* __restrict__ h) {
  int n = blockIdx.x, tid = threadIdx.x;
  if (n >= N_) {
    for (int c = tid; c < H_; c += 256) { w[n * H_ + c] = 0.0f; h[n * H_ + c] = 0.0f; }
    return;
  }
  int j = n % J_, b = n / J_;
  int s0 = slot[j * LSLOT_ + 0], s1 = slot[j * LSLOT_ + 1];
  int s2 = slot[j * LSLOT_ + 2], s3 = slot[j * LSLOT_ + 3];
  for (int c = tid; c < H_; c += 256) {
    w[n * H_ + c] = embed[(long long)s0 * H_ + c] + embed[(long long)s1 * H_ + c] +
                    embed[(long long)s2 * H_ + c] + embed[(long long)s3 * H_ + c];
    h[n * H_ + c] = hidden[b * H_ + c];
  }
}

// ---------------- GRU cell: one block per (padded) row ----------------
__global__ void k_gru(const float* __restrict__ w, const float* __restrict__ hin,
                      const float* __restrict__ Wih, const float* __restrict__ Whh,
                      const float* __restrict__ bih, const float* __restrict__ bhh,
                      float* __restrict__ hout, __bf16* __restrict__ hA) {
  __shared__ float sw[H_];
  __shared__ float sh[H_];
  int n = blockIdx.x, tid = threadIdx.x;
  if (n >= N_) {  // zero pad rows so WMMA A-fragment loads are clean
    for (int c = tid; c < H_; c += 256) { hout[n * H_ + c] = 0.0f; hA[n * H_ + c] = f2bf(0.0f); }
    return;
  }
  for (int c = tid; c < H_; c += 256) { sw[c] = w[n * H_ + c]; sh[c] = hin[n * H_ + c]; }
  __syncthreads();
  for (int i = tid; i < H_; i += 256) {
    const float* wr = Wih + (long long)i * H_;
    const float* wz = Wih + (long long)(H_ + i) * H_;
    const float* wn = Wih + (long long)(2 * H_ + i) * H_;
    const float* vr = Whh + (long long)i * H_;
    const float* vz = Whh + (long long)(H_ + i) * H_;
    const float* vn = Whh + (long long)(2 * H_ + i) * H_;
    float ir = bih[i], iz = bih[H_ + i], inn = bih[2 * H_ + i];
    float hr = bhh[i], hz = bhh[H_ + i], hn = bhh[2 * H_ + i];
    for (int k = 0; k < H_; ++k) {
      float wk = sw[k], hk = sh[k];
      ir = fmaf(wr[k], wk, ir); iz = fmaf(wz[k], wk, iz); inn = fmaf(wn[k], wk, inn);
      hr = fmaf(vr[k], hk, hr); hz = fmaf(vz[k], hk, hz); hn  = fmaf(vn[k], hk, hn);
    }
    float r = sigmoidf_(ir + hr);
    float z = sigmoidf_(iz + hz);
    float nn = tanhf(inn + r * hn);
    float hnew = (1.0f - z) * nn + z * sh[i];
    hout[n * H_ + i] = hnew;
    hA[n * H_ + i] = f2bf(hnew);
  }
}

// ---------------- attention over S, context, p_gen, (gate at t==0) ----------------
__global__ void k_attn(const float* __restrict__ enc, const int* __restrict__ masks,
                       const float* __restrict__ w, const float* __restrict__ hnew,
                       const float* __restrict__ wgenW, const float* __restrict__ wgenB,
                       const float* __restrict__ wgateW, const float* __restrict__ wgateB,
                       float* __restrict__ attn, float* __restrict__ pgen,
                       float* __restrict__ gate_out, int write_gate) {
  __shared__ float hN[H_];
  __shared__ float sc[S_];
  __shared__ float red[256];
  int n = blockIdx.x, tid = threadIdx.x;
  int b = n / J_;
  const float* encb = enc + (long long)b * S_ * H_;
  for (int c = tid; c < H_; c += 256) hN[c] = hnew[n * H_ + c];
  __syncthreads();
  // scores
  for (int s = tid; s < S_; s += 256) {
    const float* row = encb + (long long)s * H_;
    float e = 0.0f;
    for (int h = 0; h < H_; ++h) e = fmaf(row[h], hN[h], e);
    if (masks[b * S_ + s] != 1) e = -1e9f;
    sc[s] = e;
  }
  __syncthreads();
  // softmax over S
  float lm = -INFINITY;
  for (int s = tid; s < S_; s += 256) lm = fmaxf(lm, sc[s]);
  red[tid] = lm; __syncthreads();
  for (int o = 128; o > 0; o >>= 1) { if (tid < o) red[tid] = fmaxf(red[tid], red[tid + o]); __syncthreads(); }
  float mx = red[0]; __syncthreads();
  float ls = 0.0f;
  for (int s = tid; s < S_; s += 256) { float e = __expf(sc[s] - mx); sc[s] = e; ls += e; }
  red[tid] = ls; __syncthreads();
  for (int o = 128; o > 0; o >>= 1) { if (tid < o) red[tid] += red[tid + o]; __syncthreads(); }
  float inv = 1.0f / red[0]; __syncthreads();
  for (int s = tid; s < S_; s += 256) { sc[s] *= inv; attn[(long long)n * S_ + s] = sc[s]; }
  __syncthreads();
  // context (3 channels per thread) + p_gen partial
  float ctx[3];
  float pg = 0.0f;
  for (int ii = 0; ii < 3; ++ii) {
    int h = tid + ii * 256;
    float c = 0.0f;
    for (int s = 0; s < S_; ++s) c = fmaf(sc[s], encb[(long long)s * H_ + h], c);
    ctx[ii] = c;
    pg += wgenW[h] * w[n * H_ + h] + wgenW[H_ + h] * hN[h] + wgenW[2 * H_ + h] * c;
  }
  red[tid] = pg; __syncthreads();
  for (int o = 128; o > 0; o >>= 1) { if (tid < o) red[tid] += red[tid + o]; __syncthreads(); }
  if (tid == 0) pgen[n] = sigmoidf_(red[0] + wgenB[0]);
  __syncthreads();
  if (write_gate) {
    for (int g = 0; g < NG_; ++g) {
      float p = 0.0f;
      for (int ii = 0; ii < 3; ++ii) { int h = tid + ii * 256; p += wgateW[g * H_ + h] * ctx[ii]; }
      red[tid] = p; __syncthreads();
      for (int o = 128; o > 0; o >>= 1) { if (tid < o) red[tid] += red[tid + o]; __syncthreads(); }
      if (tid == 0) gate_out[n * NG_ + g] = red[0] + wgateB[g];
      __syncthreads();
    }
  }
}

// ---------------- vocab logits via bf16 WMMA: one wave per 16x16 tile ----------------
// Writes raw logits straight into the output slice out[(n*T + t)*V + v].
__global__ void k_vocab_wmma(const __bf16* __restrict__ hA, const __bf16* __restrict__ embB,
                             float* __restrict__ out, int t) {
  int v0 = blockIdx.x * 16;
  int m0 = blockIdx.y * 16;
  int lane = threadIdx.x;         // 0..31 (wave32)
  int half = lane >> 4;           // 0 or 1
  int lmod = lane & 15;
  const __bf16* arow = hA   + (long long)(m0 + lmod) * H_;   // A: row m, K along H
  const __bf16* brow = embB + (long long)(v0 + lmod) * H_;   // B col v = embed row v
  const int kbA = half * 16;   // A lane K base (contiguous, matches B)
  const int kbB = half * 16;   // B lane K base (contiguous, matches A)
  v8f acc = {};
  for (int k = 0; k < H_; k += 32) {
    const __bf16* pa = arow + k + kbA;
    v16bf a = *(const v16bf*)(pa);        // K = kbA .. kbA+15 contiguous
    v16bf bfrag = *(const v16bf*)(brow + k + kbB);  // K = kbB .. kbB+15 contiguous
    acc = __builtin_amdgcn_wmma_f32_16x16x32_bf16(
        /*neg_a=*/false, a, /*neg_b=*/false, bfrag,
        /*c_mod=*/(short)0, acc, /*reuse_a=*/false, /*reuse_b=*/false);
  }
  // C/D layout: VGPR i -> row m0 + i + 8*half, col v0 + lmod
  int v = v0 + lmod;
  if (v < V_) {
#pragma unroll
    for (int i = 0; i < 8; ++i) {
      int m = m0 + i + half * 8;
      if (m < N_) out[((long long)m * T_ + t) * V_ + v] = acc[i];
    }
  }
}

// ---------------- in-place softmax over V and scale by p_gen ----------------
__global__ void k_softmax_blend(float* __restrict__ out, const float* __restrict__ pgen, int t) {
  __shared__ float red[256];
  int n = blockIdx.x, tid = threadIdx.x;
  float* row = out + ((long long)n * T_ + t) * V_;
  float lm = -INFINITY;
  for (int v = tid; v < V_; v += 256) lm = fmaxf(lm, row[v]);
  red[tid] = lm; __syncthreads();
  for (int o = 128; o > 0; o >>= 1) { if (tid < o) red[tid] = fmaxf(red[tid], red[tid + o]); __syncthreads(); }
  float mx = red[0]; __syncthreads();
  float ls = 0.0f;
  for (int v = tid; v < V_; v += 256) ls += __expf(row[v] - mx);
  red[tid] = ls; __syncthreads();
  for (int o = 128; o > 0; o >>= 1) { if (tid < o) red[tid] += red[tid + o]; __syncthreads(); }
  float scale = pgen[n] / red[0];
  for (int v = tid; v < V_; v += 256) row[v] = scale * __expf(row[v] - mx);
}

// ---------------- pointer scatter: out[n, ids[b,s]] += (1-p_gen)*attn[n,s] ----------------
__global__ void k_scatter(float* __restrict__ out, const int* __restrict__ ids,
                          const float* __restrict__ attn, const float* __restrict__ pgen, int t) {
  int n = blockIdx.x, s = threadIdx.x;   // blockDim = S_
  int b = n / J_;
  float val = (1.0f - pgen[n]) * attn[(long long)n * S_ + s];
  int id = ids[b * S_ + s];
  atomicAdd(out + ((long long)n * T_ + t) * V_ + id, val);
}

// ---------------- argmax over V (first-index tie-break) + greedy embed feedback ----------------
__global__ void k_argmax_next(const float* __restrict__ out, const float* __restrict__ embed,
                              float* __restrict__ w, int t) {
  __shared__ float redv[256];
  __shared__ int   redi[256];
  int n = blockIdx.x, tid = threadIdx.x;
  const float* row = out + ((long long)n * T_ + t) * V_;
  float bv = -INFINITY; int bi = 0x7FFFFFFF;
  for (int v = tid; v < V_; v += 256) {
    float x = row[v];
    if (x > bv || (x == bv && v < bi)) { bv = x; bi = v; }
  }
  redv[tid] = bv; redi[tid] = bi; __syncthreads();
  for (int o = 128; o > 0; o >>= 1) {
    if (tid < o) {
      float xv = redv[tid + o]; int xi = redi[tid + o];
      if (xv > redv[tid] || (xv == redv[tid] && xi < redi[tid])) { redv[tid] = xv; redi[tid] = xi; }
    }
    __syncthreads();
  }
  int idx = redi[0];
  for (int c = tid; c < H_; c += 256) w[n * H_ + c] = embed[(long long)idx * H_ + c];
}

// ---------------- host driver ----------------
extern "C" void kernel_launch(void* const* d_in, const int* in_sizes, int n_in,
                              void* d_out, int out_size, void* d_ws, size_t ws_size,
                              hipStream_t stream) {
  (void)in_sizes; (void)n_in; (void)out_size; (void)ws_size;
  const int*   ids    = (const int*)d_in[0];
  const float* enc    = (const float*)d_in[1];
  const float* hid    = (const float*)d_in[2];
  const int*   masks  = (const int*)d_in[3];
  const int*   slot   = (const int*)d_in[4];
  /* d_in[5] = max_len (compile-time T_=5) */
  const float* embed  = (const float*)d_in[6];
  const float* Wih    = (const float*)d_in[7];
  const float* Whh    = (const float*)d_in[8];
  const float* bih    = (const float*)d_in[9];
  const float* bhh    = (const float*)d_in[10];
  const float* wgenW  = (const float*)d_in[11];
  const float* wgenB  = (const float*)d_in[12];
  const float* wgateW = (const float*)d_in[13];
  const float* wgateB = (const float*)d_in[14];

  float* out = (float*)d_out;
  float* gate_out = out + (long long)N_ * T_ * V_;   // all_point first, then all_gate

  char* ws = (char*)d_ws;
  size_t off = 0;
  auto wsalloc = [&](size_t bytes) -> char* {
    char* p = ws + off;
    off += (bytes + 255) & ~(size_t)255;
    return p;
  };
  __bf16* embB = (__bf16*)wsalloc((size_t)VP_ * H_ * sizeof(__bf16));  // ~53.8 MB, L2-resident
  __bf16* hA   = (__bf16*)wsalloc((size_t)MP_ * H_ * sizeof(__bf16));
  float*  h0   = (float*)wsalloc((size_t)MP_ * H_ * sizeof(float));
  float*  h1   = (float*)wsalloc((size_t)MP_ * H_ * sizeof(float));
  float*  wbuf = (float*)wsalloc((size_t)MP_ * H_ * sizeof(float));
  float*  attn = (float*)wsalloc((size_t)N_ * S_ * sizeof(float));
  float*  pgen = (float*)wsalloc((size_t)MP_ * sizeof(float));

  {
    long long tot = (long long)VP_ * H_;
    int blocks = (int)((tot + 255) / 256);
    k_embed_bf16<<<blocks, 256, 0, stream>>>(embed, embB);
  }
  k_init_state<<<MP_, 256, 0, stream>>>(embed, hid, slot, wbuf, h0);

  float* hc = h0;
  float* hn = h1;
  for (int t = 0; t < T_; ++t) {
    k_gru<<<MP_, 256, 0, stream>>>(wbuf, hc, Wih, Whh, bih, bhh, hn, hA);
    k_attn<<<N_, 256, 0, stream>>>(enc, masks, wbuf, hn, wgenW, wgenB, wgateW, wgateB,
                                   attn, pgen, gate_out, (t == 0) ? 1 : 0);
    dim3 grid(VP_ / 16, MP_ / 16);   // 2188 x 12 tiles, 1 wave each
    k_vocab_wmma<<<grid, 32, 0, stream>>>(hA, embB, out, t);
    k_softmax_blend<<<N_, 256, 0, stream>>>(out, pgen, t);
    k_scatter<<<N_, S_, 0, stream>>>(out, ids, attn, pgen, t);
    if (t + 1 < T_) k_argmax_next<<<N_, 256, 0, stream>>>(out, embed, wbuf, t);
    float* tmp = hc; hc = hn; hn = tmp;
  }
}
